// BaselineSelfAttention_37203006718040
// MI455X (gfx1250) — matrix operand order repaired and checked by reading, not compiled, not run on hardware
//
#include <hip/hip_runtime.h>
#include <hip/hip_bf16.h>

// ---------------------------------------------------------------------------
// Causal MHA forward for MI455X (gfx1250, wave32, WMMA bf16->f32 accumulate).
// B=2, S=4096, D=768, H=12, HD=64.
// GEMMs stage weight tiles into LDS via the Tensor Data Mover (TDM),
// double-buffered, synchronized with s_wait_tensorcnt + workgroup barriers.
// ---------------------------------------------------------------------------

typedef __bf16 bf16_t;
typedef __attribute__((ext_vector_type(16))) __bf16 v16bf;
typedef __attribute__((ext_vector_type(8)))  float  v8f;
typedef __attribute__((ext_vector_type(4)))  unsigned int v4u;
typedef __attribute__((ext_vector_type(8)))  int v8i;
typedef __attribute__((ext_vector_type(4)))  int v4i;

#define B_   2
#define S_   4096
#define D_   768
#define H_   12
#define HD_  64
#define BS_  (B_ * S_)          // 8192 rows
#define KSTEPS_ (D_ / 32)       // 24 k-steps in the GEMM mainloop

// ---------------- fragment helpers (CDNA5 16-bit WMMA layouts) -------------
// A-matrix 16x32 bf16: lane m = lane&15; this code puts K = 16*half + [0..8) in VGPR0..3,
// K = 16*half + 8 + [0..8) in VGPR4..7 (k = 16*half + s, as B).  -> two contiguous 16B loads.
__device__ __forceinline__ v16bf ld_a_2x16(const bf16_t* p_lo, const bf16_t* p_hi) {
    v16bf r;
    reinterpret_cast<float4*>(&r)[0] = *reinterpret_cast<const float4*>(p_lo);
    reinterpret_cast<float4*>(&r)[1] = *reinterpret_cast<const float4*>(p_hi);
    return r;
}

// B-matrix 32x16 bf16: lane n = lane&15, khalf = lane>>4; VGPR v holds
// K = 16*khalf + 2v, 2v+1  -> one contiguous 32B load per lane.
__device__ __forceinline__ v16bf ld_b_32(const bf16_t* p) {
    v16bf r;
    reinterpret_cast<float4*>(&r)[0] = reinterpret_cast<const float4*>(p)[0];
    reinterpret_cast<float4*>(&r)[1] = reinterpret_cast<const float4*>(p)[1];
    return r;
}

__device__ __forceinline__ v8f wmma_bf16(v16bf a, v16bf b, v8f c) {
    return __builtin_amdgcn_wmma_f32_16x16x32_bf16(
        /*neg_a=*/false, a, /*neg_b=*/false, b,
        /*c_mod=*/(short)0, c, /*reuse_a=*/false, /*reuse_b=*/false);
}

__device__ __forceinline__ float redmax16(float v) {
    v = fmaxf(v, __shfl_xor(v, 1));
    v = fmaxf(v, __shfl_xor(v, 2));
    v = fmaxf(v, __shfl_xor(v, 4));
    v = fmaxf(v, __shfl_xor(v, 8));
    return v;
}
__device__ __forceinline__ float redsum16(float v) {
    v += __shfl_xor(v, 1);
    v += __shfl_xor(v, 2);
    v += __shfl_xor(v, 4);
    v += __shfl_xor(v, 8);
    return v;
}

// ---------------- TDM: DMA a 64(N) x 32(K) bf16 tile into LDS --------------
// D# per CDNA5 ISA ch.8: group0 = {count/flags, lds_addr, global_addr, type},
// group1 = {mask/data_size, tensor dims, tile dims, dim0 stride}. 2-D tensor,
// groups 2/3 unused (zero). data_size=1 (2 bytes). tensor dims == tile dims
// so the transfer is always in-bounds. dim0 stride = 768 elements (W rows).
__device__ __forceinline__ void tdm_load_w_tile(const bf16_t* gtile, unsigned int lds_off) {
    unsigned long long ga = (unsigned long long)(uintptr_t)gtile;
    v4u g0;
    g0[0] = 1u;                                             // count=1, no gather
    g0[1] = lds_off;                                        // lds byte address
    g0[2] = (unsigned int)(ga & 0xFFFFFFFFu);               // global_addr[31:0]
    g0[3] = (unsigned int)((ga >> 32) & 0x01FFFFFFu) | (2u << 30); // addr[56:32] | type=2
    v8i g1;
    g1[0] = 0x00010000;            // workgroup_mask=0, data_size=1 (2B), no flags
    g1[1] = (int)(32u << 16);      // tensor_dim0 = 32 (bits 79:48, low 16 here)
    g1[2] = (int)(64u << 16);      // tensor_dim0 hi | tensor_dim1 = 64 (low 16)
    g1[3] = (int)(32u << 16);      // tensor_dim1 hi | tile_dim0 = 32
    g1[4] = 64;                    // tile_dim1 = 64, tile_dim2 = 0
    g1[5] = D_;                    // tensor_dim0_stride = 768 (elements)
    g1[6] = 0;                     // stride0 hi | stride1 lo (unused, 2-D)
    g1[7] = 0;
    v4i z4 = {0, 0, 0, 0};
    v8i z8 = {0, 0, 0, 0, 0, 0, 0, 0};
    __builtin_amdgcn_tensor_load_to_lds(g0, g1, z4, z4, z8, 0);
}

__device__ __forceinline__ unsigned int lds_offset_of(const void* p) {
    // LDS aperture keeps the LDS byte offset in addr[31:0] of the flat address.
    return (unsigned int)(uintptr_t)p;
}

// ---------------- fp32 -> bf16 conversion ----------------------------------
__global__ void cvt_f32_bf16(const float* __restrict__ src,
                             bf16_t* __restrict__ dst, int n) {
    int i = blockIdx.x * blockDim.x + threadIdx.x;
    if (i < n) dst[i] = (bf16_t)src[i];
}

// ---------------- block GEMM mainloop: 128(M) x 64(N), K = 768 -------------
// 8 waves / block, each wave owns 16 rows. Weight tile (64x32, shared by all
// waves) is TDM-staged into double-buffered LDS; wave 0 drives the pipeline.
// A-fragments stream from global (rows are block-private, L2-resident).
__device__ __forceinline__ void gemm_block_tdm(
    const bf16_t* __restrict__ A, int row0_wave,
    const bf16_t* __restrict__ W, int col0,
    bf16_t* __restrict__ btile /* [2][64*32] */, v8f acc[4])
{
    const int wave = threadIdx.x >> 5;
    const int lane = threadIdx.x & 31;
    const int n16  = lane & 15;
    const int half = lane >> 4;
    const bf16_t* arow  = A + (size_t)(row0_wave + n16) * D_;   // A lane row m = lane&15
    const bf16_t* wbase = W + (size_t)col0 * D_;
    const unsigned int lds0 = lds_offset_of(btile);

    if (wave == 0) tdm_load_w_tile(wbase, lds0);                 // prime buffer 0

    for (int i = 0; i < KSTEPS_; ++i) {
        const int kb = i * 32;
        if (wave == 0) {
            if (i + 1 < KSTEPS_) {
                tdm_load_w_tile(wbase + kb + 32, lds0 + ((i + 1) & 1) * (64 * 32 * 2));
                __builtin_amdgcn_s_wait_tensorcnt(1);    // buffer i done, i+1 in flight
            } else {
                __builtin_amdgcn_s_wait_tensorcnt(0);    // drain on last iteration
            }
        }
        __syncthreads();                                         // buffer i ready

        v16bf a = ld_a_2x16(arow + kb + 16 * half, arow + kb + 8 + 16 * half);
        const bf16_t* bt = btile + (i & 1) * (64 * 32);
#pragma unroll
        for (int f = 0; f < 4; ++f) {
            const bf16_t* bp = bt + (f * 16 + n16) * 32 + 16 * half;
            acc[f] = wmma_bf16(a, ld_b_32(bp), acc[f]);
        }
        __syncthreads();                     // all waves done before buffer reuse
    }
}

// ---------------- QKV projection -------------------------------------------
// grid (64, 12, 3), block 256 (8 waves). wave -> 16 rows, head h -> 64 cols.
// Q,K stored [b,h,s,64]; V stored transposed [b,h,64,s].
__global__ __launch_bounds__(256) void proj_qkv(
    const bf16_t* __restrict__ xb,
    const bf16_t* __restrict__ wqb, const bf16_t* __restrict__ wkb,
    const bf16_t* __restrict__ wvb,
    bf16_t* __restrict__ Q, bf16_t* __restrict__ K, bf16_t* __restrict__ Vt)
{
    __shared__ bf16_t btile[2][64 * 32];

    const int wave = threadIdx.x >> 5;
    const int lane = threadIdx.x & 31;
    const int n16  = lane & 15;
    const int half = lane >> 4;
    const int row0 = (blockIdx.x * 8 + wave) * 16;    // [0, 8192)
    const int h    = blockIdx.y;
    const int which = blockIdx.z;
    const bf16_t* W = (which == 0) ? wqb : (which == 1) ? wkb : wvb;

    v8f acc[4];
#pragma unroll
    for (int f = 0; f < 4; ++f)
#pragma unroll
        for (int r = 0; r < 8; ++r) acc[f][r] = 0.0f;

    gemm_block_tdm(xb, row0, W, h * HD_, &btile[0][0], acc);

    const int b  = row0 >> 12;        // row0 / 4096
    const int s0 = row0 & (S_ - 1);
    const size_t bh = (size_t)(b * H_ + h);

    if (which < 2) {
        bf16_t* dst = ((which == 0) ? Q : K) + bh * S_ * HD_;
#pragma unroll
        for (int f = 0; f < 4; ++f)
#pragma unroll
            for (int r = 0; r < 8; ++r) {
                int m = r + 8 * half;
                dst[(size_t)(s0 + m) * HD_ + f * 16 + n16] = (bf16_t)acc[f][r];
            }
    } else {
        bf16_t* dst = Vt + bh * HD_ * S_;
#pragma unroll
        for (int f = 0; f < 4; ++f)
#pragma unroll
            for (int r = 0; r < 8; ++r) {
                int m = r + 8 * half;
                dst[(size_t)(f * 16 + n16) * S_ + s0 + m] = (bf16_t)acc[f][r];
            }
    }
}

// ---------------- flash attention (causal, online softmax) -----------------
// grid (24, 64), block 128 (4 waves). Each wave owns a 16-row query tile and
// streams 32-wide key tiles. attn written bf16 [b, s, h*64+hd] = [BS, 768].
__global__ __launch_bounds__(128) void flash_attn(
    const bf16_t* __restrict__ Q, const bf16_t* __restrict__ K,
    const bf16_t* __restrict__ Vt, bf16_t* __restrict__ attn)
{
    __shared__ bf16_t lds_p[4][16 * 32];      // per-wave P scratch (re-layout)

    const int wave = threadIdx.x >> 5;
    const int lane = threadIdx.x & 31;
    const int n16  = lane & 15;
    const int half = lane >> 4;
    const int bh   = blockIdx.x;                       // b*12 + h
    const int q0   = (blockIdx.y * 4 + wave) * 16;     // query tile start

    const bf16_t* Qb = Q  + (size_t)bh * S_ * HD_;
    const bf16_t* Kb = K  + (size_t)bh * S_ * HD_;
    const bf16_t* Vb = Vt + (size_t)bh * HD_ * S_;

    // Q A-fragments (16 x 64 split into two 16x32 k-steps), loaded once.
    const bf16_t* qrow = Qb + (size_t)(q0 + n16) * HD_;
    v16bf qa0 = ld_a_2x16(qrow + 16 * half,      qrow + 8 + 16 * half);
    v16bf qa1 = ld_a_2x16(qrow + 32 + 16 * half, qrow + 40 + 16 * half);

    float mi[8], li[8];
    v8f o[4];
#pragma unroll
    for (int r = 0; r < 8; ++r) { mi[r] = -1e30f; li[r] = 0.0f; }
#pragma unroll
    for (int f = 0; f < 4; ++f)
#pragma unroll
        for (int r = 0; r < 8; ++r) o[f][r] = 0.0f;

    bf16_t* pbuf = &lds_p[wave][0];
    const float scale = 0.125f;                // 1/sqrt(64)

    for (int j = 0; j < q0 + 16; j += 32) {
        // ---- scores: S[16 x 32] = Q_tile (16x64) . K_tile^T --------------
        v8f c0, c1;
#pragma unroll
        for (int r = 0; r < 8; ++r) { c0[r] = 0.0f; c1[r] = 0.0f; }
        {
            const bf16_t* k0 = Kb + (size_t)(j + n16) * HD_ + 16 * half;
            c0 = wmma_bf16(qa0, ld_b_32(k0),      c0);
            c0 = wmma_bf16(qa1, ld_b_32(k0 + 32), c0);
            const bf16_t* k1 = Kb + (size_t)(j + 16 + n16) * HD_ + 16 * half;
            c1 = wmma_bf16(qa0, ld_b_32(k1),      c1);
            c1 = wmma_bf16(qa1, ld_b_32(k1 + 32), c1);
        }

        // ---- mask + scale + online softmax (rows m = r + 8*half) ---------
#pragma unroll
        for (int r = 0; r < 8; ++r) {
            const int m = r + 8 * half;
            float s0v = (j + n16      <= q0 + m) ? c0[r] * scale : -1e30f;
            float s1v = (j + 16 + n16 <= q0 + m) ? c1[r] * scale : -1e30f;
            float mnew = fmaxf(mi[r], redmax16(fmaxf(s0v, s1v)));
            float p0 = __expf(s0v - mnew);
            float p1 = __expf(s1v - mnew);
            float corr = __expf(mi[r] - mnew);
            li[r] = li[r] * corr + redsum16(p0 + p1);
            mi[r] = mnew;
            o[0][r] *= corr; o[1][r] *= corr; o[2][r] *= corr; o[3][r] *= corr;
            // P -> LDS (row-major 16x32 bf16) for C-frag -> A-frag re-layout
            pbuf[m * 32 + n16]      = (bf16_t)p0;
            pbuf[m * 32 + 16 + n16] = (bf16_t)p1;
        }
        // per-wave scratch; DS ops are in-order within a wave.

        // ---- O += P (16x32) . V_tile (32x64) -----------------------------
        v16bf pa = ld_a_2x16(pbuf + n16 * 32 + 16 * half,
                             pbuf + n16 * 32 + 8 + 16 * half);
#pragma unroll
        for (int f = 0; f < 4; ++f) {
            const bf16_t* vrow = Vb + (size_t)(f * 16 + n16) * S_ + j + 16 * half;
            o[f] = wmma_bf16(pa, ld_b_32(vrow), o[f]);
        }
    }

    // ---- normalize and store bf16 attn[b, s, h*64 + hd] ------------------
    const int b = bh / H_;
    const int h = bh % H_;
    bf16_t* arow = attn + ((size_t)(b * S_ + q0)) * D_ + h * HD_;
#pragma unroll
    for (int f = 0; f < 4; ++f)
#pragma unroll
        for (int r = 0; r < 8; ++r) {
            int m = r + 8 * half;
            arow[(size_t)m * D_ + f * 16 + n16] = (bf16_t)(o[f][r] / li[r]);
        }
}

// ---------------- output projection ----------------------------------------
// grid (64, 12), block 256. out = attn @ Wo^T, fp32 result.
__global__ __launch_bounds__(256) void out_proj(
    const bf16_t* __restrict__ attnb, const bf16_t* __restrict__ wob,
    float* __restrict__ out)
{
    __shared__ bf16_t btile[2][64 * 32];

    const int wave = threadIdx.x >> 5;
    const int lane = threadIdx.x & 31;
    const int n16  = lane & 15;
    const int half = lane >> 4;
    const int row0 = (blockIdx.x * 8 + wave) * 16;
    const int col0 = blockIdx.y * 64;

    v8f acc[4];
#pragma unroll
    for (int f = 0; f < 4; ++f)
#pragma unroll
        for (int r = 0; r < 8; ++r) acc[f][r] = 0.0f;

    gemm_block_tdm(attnb, row0, wob, col0, &btile[0][0], acc);

#pragma unroll
    for (int f = 0; f < 4; ++f)
#pragma unroll
        for (int r = 0; r < 8; ++r) {
            int m = r + 8 * half;
            out[(size_t)(row0 + m) * D_ + col0 + f * 16 + n16] = acc[f][r];
        }
}

// ---------------------------------------------------------------------------
extern "C" void kernel_launch(void* const* d_in, const int* in_sizes, int n_in,
                              void* d_out, int out_size, void* d_ws, size_t ws_size,
                              hipStream_t stream) {
    const float* x  = (const float*)d_in[0];
    const float* wq = (const float*)d_in[1];
    const float* wk = (const float*)d_in[2];
    const float* wv = (const float*)d_in[3];
    const float* wo = (const float*)d_in[4];
    float* out = (float*)d_out;

    const size_t nx = (size_t)BS_ * D_;       // 6,291,456
    const size_t nw = (size_t)D_ * D_;        //   589,824

    char* ws = (char*)d_ws;
    size_t off = 0;
    bf16_t* xb   = (bf16_t*)(ws + off); off += nx * sizeof(bf16_t);
    bf16_t* wqb  = (bf16_t*)(ws + off); off += nw * sizeof(bf16_t);
    bf16_t* wkb  = (bf16_t*)(ws + off); off += nw * sizeof(bf16_t);
    bf16_t* wvb  = (bf16_t*)(ws + off); off += nw * sizeof(bf16_t);
    bf16_t* wob  = (bf16_t*)(ws + off); off += nw * sizeof(bf16_t);
    bf16_t* Qb   = (bf16_t*)(ws + off); off += nx * sizeof(bf16_t);
    bf16_t* Kb   = (bf16_t*)(ws + off); off += nx * sizeof(bf16_t);
    bf16_t* Vtb  = (bf16_t*)(ws + off); off += nx * sizeof(bf16_t);
    bf16_t* attn = (bf16_t*)(ws + off); off += nx * sizeof(bf16_t);

    // 1) fp32 -> bf16 conversions
    cvt_f32_bf16<<<(int)((nx + 255) / 256), 256, 0, stream>>>(x,  xb,  (int)nx);
    cvt_f32_bf16<<<(int)((nw + 255) / 256), 256, 0, stream>>>(wq, wqb, (int)nw);
    cvt_f32_bf16<<<(int)((nw + 255) / 256), 256, 0, stream>>>(wk, wkb, (int)nw);
    cvt_f32_bf16<<<(int)((nw + 255) / 256), 256, 0, stream>>>(wv, wvb, (int)nw);
    cvt_f32_bf16<<<(int)((nw + 255) / 256), 256, 0, stream>>>(wo, wob, (int)nw);

    // 2) QKV projections (WMMA + TDM-staged weights)
    proj_qkv<<<dim3(BS_ / 128, H_, 3), 256, 0, stream>>>(xb, wqb, wkb, wvb,
                                                         Qb, Kb, Vtb);

    // 3) causal flash attention (WMMA + online softmax)
    flash_attn<<<dim3(B_ * H_, S_ / (16 * 4)), 128, 0, stream>>>(Qb, Kb, Vtb, attn);

    // 4) output projection (WMMA + TDM-staged weights, fp32 out)
    out_proj<<<dim3(BS_ / 128, D_ / 64), 256, 0, stream>>>(attn, wob, out);
}
